// OptimizedCueBasedRNNModel_MH_6262062317724
// MI455X (gfx1250) — compile-verified, matrix operand order checked by reading, not run on hardware
//
#include <hip/hip_runtime.h>
#include <hip/hip_bf16.h>

typedef __attribute__((ext_vector_type(16))) _Float16 v16h;
typedef __attribute__((ext_vector_type(8)))  _Float16 v8h;
typedef __attribute__((ext_vector_type(8)))  float    v8f;

#define T_STEPS 256
#define BATCH   32
#define HID     512
#define VOCAB   10001
#define NPAD    10016          // vocab padded to 16
#define QK_SCALE 0.125f        // 1/sqrt(64)
#define BIGSPLIT (1 << 30)

// ---------------------------------------------------------------------------
// CDNA5 async global->LDS copy (CDNA5 ISA, VGLOBAL opcode 98).
// ldsOff is the byte address inside this workgroup's LDS allocation; gptr is a
// 64-bit global address. Tracked by ASYNCcnt -> s_wait_asynccnt before use.
// ---------------------------------------------------------------------------
__device__ __forceinline__ void async_g2lds_b128(unsigned ldsOff, const void* gptr) {
  asm volatile("global_load_async_to_lds_b128 %0, %1, off"
               :: "v"(ldsOff), "v"(gptr) : "memory");
}
__device__ __forceinline__ void wait_async0() {
  asm volatile("s_wait_asynccnt 0x0" ::: "memory");
}

// 4-op tanh; saturates correctly for |x| large (exp->inf/0).
__device__ __forceinline__ float fast_tanh(float x) {
  float e = __expf(2.f * x);
  return 1.f - 2.f / (e + 1.f);
}

// ============================================================================
// Pack an f32 weight W[N][K] (used as x @ W.T) into the CDNA5 WMMA B-operand
// fragment layout for v_wmma_f32_16x16x32_f16 (ISA 7.12.2):
//   lane n (0..15) holds column N=n; halves 0..7 hold K=ks..ks+7,
//   halves 8..15 hold K=ks+16..ks+23 with ks = (lane>=16 ? 8 : 0).
// Tile-linear storage: [(kt*Ntiles + nt)*32 + lane]*16 halves.
// ============================================================================
__global__ void pack_b_frag(const float* __restrict__ W, _Float16* __restrict__ out,
                            int Ntiles, int Ktiles, int K, int Nsrc) {
  int g    = blockIdx.x * blockDim.x + threadIdx.x;
  int lane = g & 31;
  int tile = g >> 5;
  if (tile >= Ntiles * Ktiles) return;
  int kt = tile / Ntiles;
  int nt = tile - kt * Ntiles;
  int n  = nt * 16 + (lane & 15);
  int ks = (lane >> 4) * 8;
  v16h h;
#pragma unroll
  for (int j = 0; j < 8; ++j) {
    int k0 = kt * 32 + ks + j;
    h[j]     = (n < Nsrc) ? (_Float16)W[(size_t)n * K + k0]      : (_Float16)0.f;
    h[8 + j] = (n < Nsrc) ? (_Float16)W[(size_t)n * K + k0 + 16] : (_Float16)0.f;
  }
  *(v16h*)(out + (size_t)tile * 512 + lane * 16) = h;
}

__device__ __forceinline__ v16h load_a_frag(const _Float16* ap) {
  v8h lo = *(const v8h*)ap;            // K = base..base+7
  v8h hi = *(const v8h*)(ap + 16);     // K = base+16..base+23
  v16h a;
#pragma unroll
  for (int j = 0; j < 8; ++j) { a[j] = lo[j]; a[8 + j] = hi[j]; }
  return a;
}

// ============================================================================
// 2x2-blocked wave GEMM: each wave owns BOTH 16-row M-tiles and a pair of
// N-tiles -> 4 accumulators; A/B each loaded once, used twice per WMMA pair.
// kt-loop unroll capped to avoid scratch spills.
// ============================================================================
__device__ __forceinline__ void wave_gemm_f16(
    const _Float16* __restrict__ act, int actStride,            // halves
    int splitAt, int splitShift,
    const _Float16* __restrict__ packedW, int Ntiles, int Ktiles,
    const float* __restrict__ bias,
    _Float16* __restrict__ out, int outStride,                  // halves
    int wave, int nwaves, int lane) {
  int nPairs = Ntiles >> 1;
  int rlo = lane & 15;
  int ks  = (lane >> 4) * 8;
  for (int pidx = wave; pidx < nPairs; pidx += nwaves) {
    int nt0 = pidx * 2;
    v8f c00 = {}, c01 = {}, c10 = {}, c11 = {};
    const _Float16* bp0 = packedW + (size_t)nt0 * 512 + lane * 16;
#pragma unroll 2
    for (int kt = 0; kt < Ktiles; ++kt) {
      int col = kt * 32;
      if (col >= splitAt) col += splitShift;
      const _Float16* ap0 = act + (size_t)rlo * actStride + col + ks;
      v16h a0 = load_a_frag(ap0);                          // rows 0..15
      v16h a1 = load_a_frag(ap0 + (size_t)16 * actStride); // rows 16..31
      size_t boff = (size_t)kt * Ntiles * 512;
      v16h b0 = *(const v16h*)(bp0 + boff);
      v16h b1 = *(const v16h*)(bp0 + boff + 512);
      c00 = __builtin_amdgcn_wmma_f32_16x16x32_f16(false, a0, false, b0, (short)0, c00, false, false);
      c01 = __builtin_amdgcn_wmma_f32_16x16x32_f16(false, a0, false, b1, (short)0, c01, false, false);
      c10 = __builtin_amdgcn_wmma_f32_16x16x32_f16(false, a1, false, b0, (short)0, c10, false, false);
      c11 = __builtin_amdgcn_wmma_f32_16x16x32_f16(false, a1, false, b1, (short)0, c11, false, false);
    }
    // C/D layout: lanes 0-15 -> M=vgpr, lanes 16-31 -> M=vgpr+8; N = lane&15
    int subrow = (lane >= 16) ? 8 : 0;
    int nc0 = nt0 * 16 + (lane & 15);
    int nc1 = nc0 + 16;
    float bv0 = bias ? bias[nc0] : 0.f;
    float bv1 = bias ? bias[nc1] : 0.f;
#pragma unroll
    for (int v = 0; v < 8; ++v) {
      out[(size_t)(subrow + v) * outStride + nc0]      = (_Float16)(c00[v] + bv0);
      out[(size_t)(subrow + v) * outStride + nc1]      = (_Float16)(c01[v] + bv1);
      out[(size_t)(16 + subrow + v) * outStride + nc0] = (_Float16)(c10[v] + bv0);
      out[(size_t)(16 + subrow + v) * outStride + nc1] = (_Float16)(c11[v] + bv1);
    }
  }
}

// ============================================================================
// Row-wise LayerNorm + tanh over f16 [32 x ncols], 512 threads, 128-bit LDS IO.
// ncols must be a multiple of 128 (512 / 1536 / 2048 all qualify).
// ============================================================================
__device__ __forceinline__ void ln_tanh(
    const _Float16* src, int ss, int ncols,
    const float* __restrict__ g, const float* __restrict__ bb,
    _Float16* dst, int ds,
    float* pred, float* stats, int tid) {
  int r = tid >> 4, j = tid & 15;
  int chunks = ncols >> 3;                 // 8-half chunks per row
  float s = 0.f, sq = 0.f;
  for (int c8 = j; c8 < chunks; c8 += 16) {
    v8h x8 = *(const v8h*)(src + (size_t)r * ss + c8 * 8);
#pragma unroll
    for (int jj = 0; jj < 8; ++jj) {
      float x = (float)x8[jj];
      s += x; sq += x * x;
    }
  }
  pred[(r * 16 + j) * 2]     = s;
  pred[(r * 16 + j) * 2 + 1] = sq;
  __syncthreads();
  if (tid < 32) {
    float S = 0.f, Q = 0.f;
#pragma unroll
    for (int p = 0; p < 16; ++p) {
      S += pred[(tid * 16 + p) * 2];
      Q += pred[(tid * 16 + p) * 2 + 1];
    }
    float m   = S / (float)ncols;
    float var = Q / (float)ncols - m * m;
    stats[tid * 2]     = m;
    stats[tid * 2 + 1] = rsqrtf(var + 1e-5f);
  }
  __syncthreads();
  float m = stats[r * 2], rs = stats[r * 2 + 1];
  for (int c8 = j; c8 < chunks; c8 += 16) {
    v8h x8 = *(const v8h*)(src + (size_t)r * ss + c8 * 8);
    const float4* g4 = (const float4*)(g  + c8 * 8);
    const float4* b4 = (const float4*)(bb + c8 * 8);
    float4 g0 = g4[0], g1 = g4[1], b0 = b4[0], b1 = b4[1];
    float gv[8] = {g0.x, g0.y, g0.z, g0.w, g1.x, g1.y, g1.z, g1.w};
    float bv[8] = {b0.x, b0.y, b0.z, b0.w, b1.x, b1.y, b1.z, b1.w};
    v8h o;
#pragma unroll
    for (int jj = 0; jj < 8; ++jj) {
      float x = (float)x8[jj];
      o[jj] = (_Float16)fast_tanh((x - m) * rs * gv[jj] + bv[jj]);
    }
    *(v8h*)(dst + (size_t)r * ds + c8 * 8) = o;
  }
  __syncthreads();
}

// ============================================================================
// Persistent recurrent scan: one workgroup, 512 threads (16 wave32s), big LDS.
// actbuf row layout: [ emb(0:512) | query(512:1024) | attended(1024:1536) | prev_h(1536:2048) ]
// ============================================================================
__global__ void __launch_bounds__(512, 1) rnn_scan_kernel(
    const int*   __restrict__ obs,
    const float* __restrict__ hidden_init,
    const float* __restrict__ emb,
    const _Float16* __restrict__ qWp,
    const float* __restrict__ q_b,
    const float* __restrict__ qn_g, const float* __restrict__ qn_b,
    const _Float16* __restrict__ wqP,
    const _Float16* __restrict__ wkP,
    const _Float16* __restrict__ wvP,
    const _Float16* __restrict__ woP,
    const _Float16* __restrict__ intWp,
    const float* __restrict__ int_b,
    const float* __restrict__ in_g, const float* __restrict__ in_b,
    const _Float16* __restrict__ finWp,
    const float* __restrict__ fin_b,
    const float* __restrict__ fn_g, const float* __restrict__ fn_b,
    _Float16* __restrict__ Kc,       // [T][B][512] f16 cache (L2-resident, 8MB)
    _Float16* __restrict__ Vc,
    _Float16* __restrict__ hs) {     // [T][B][512] f16 outputs for decode
  extern __shared__ char smem[];
  _Float16* actbuf   = (_Float16*)smem;            // 32*2048 f16 = 128KB
  _Float16* interbuf = actbuf + 32 * 2048;         // 32*2048 f16 = 128KB
  _Float16* qhbuf    = interbuf + 32 * 2048;       // 32*512  f16 =  32KB
  float*    pred     = (float*)(qhbuf + 32 * 512); // 32*16*2 f32 =   4KB
  float*    stats    = pred + 32 * 16 * 2;         // 32*2    f32

  const int tid  = threadIdx.x;
  const int lane = tid & 31;
  const int wave = tid >> 5;
  const int NW   = 16;

  // prev_h <- hidden_init (vectorized: 8-half chunks)
  for (int idx = tid; idx < 32 * 64; idx += 512) {
    int b = idx >> 6, c8 = idx & 63;
    const float4* hf = (const float4*)(hidden_init + b * 512 + c8 * 8);
    float4 f0 = hf[0], f1 = hf[1];
    v8h o;
    o[0] = (_Float16)f0.x; o[1] = (_Float16)f0.y; o[2] = (_Float16)f0.z; o[3] = (_Float16)f0.w;
    o[4] = (_Float16)f1.x; o[5] = (_Float16)f1.y; o[6] = (_Float16)f1.z; o[7] = (_Float16)f1.w;
    *(v8h*)(actbuf + b * 2048 + 1536 + c8 * 8) = o;
  }
  __syncthreads();

  for (int i = 0; i < T_STEPS; ++i) {
    // emb gather -> actbuf[:, 0:512]  (float4 loads, v8h stores)
    for (int idx = tid; idx < 32 * 64; idx += 512) {
      int b = idx >> 6, c8 = idx & 63;
      int tok = obs[i * 32 + b];
      const float4* ef = (const float4*)(emb + (size_t)tok * 512 + c8 * 8);
      float4 f0 = ef[0], f1 = ef[1];
      v8h o;
      o[0] = (_Float16)f0.x; o[1] = (_Float16)f0.y; o[2] = (_Float16)f0.z; o[3] = (_Float16)f0.w;
      o[4] = (_Float16)f1.x; o[5] = (_Float16)f1.y; o[6] = (_Float16)f1.z; o[7] = (_Float16)f1.w;
      *(v8h*)(actbuf + b * 2048 + c8 * 8) = o;
    }
    __syncthreads();

    // GEMM1: qin[32x1024] @ q_W.T -> raw in interbuf[:, 0:512]
    wave_gemm_f16(actbuf, 2048, /*split*/512, /*shift*/1024,
                  qWp, 32, 32, q_b, interbuf, 2048, wave, NW, lane);
    __syncthreads();
    // query = tanh(LN(raw)) -> actbuf[:, 512:1024]
    ln_tanh(interbuf, 2048, 512, qn_g, qn_b, actbuf + 512, 2048, pred, stats, tid);

    // GEMM2: qh / kh / vh (K=512 each)
    wave_gemm_f16(actbuf + 512, 2048, BIGSPLIT, 0, wqP, 32, 16, nullptr,
                  qhbuf, 512, wave, NW, lane);
    wave_gemm_f16(actbuf + 512, 2048, BIGSPLIT, 0, wkP, 32, 16, nullptr,
                  Kc + (size_t)i * 32 * 512, 512, wave, NW, lane);
    wave_gemm_f16(actbuf + 512, 2048, BIGSPLIT, 0, wvP, 32, 16, nullptr,
                  Vc + (size_t)i * 32 * 512, 512, wave, NW, lane);
    __threadfence_block();
    __syncthreads();

    // Flash-style causal attention: one thread per (b,head), online softmax,
    // 128-bit K/V loads, prefetch of next-t rows (global_prefetch_b8).
    if (tid < 256) {
      int b = tid >> 3, h = tid & 7;
      const v8h* q8 = (const v8h*)(qhbuf + b * 512 + h * 64);
      float q[64];
#pragma unroll
      for (int c = 0; c < 8; ++c) {
        v8h qq = q8[c];
#pragma unroll
        for (int j = 0; j < 8; ++j) q[c * 8 + j] = (float)qq[j];
      }
      float m = -1e30f, l = 0.f;
      float acc[64];
#pragma unroll
      for (int d = 0; d < 64; ++d) acc[d] = 0.f;
      int tEnd = (i == 0) ? 1 : i;               // valid: t<i, or t==i==0
      for (int t = 0; t < tEnd; ++t) {
        const v8h* kp8 = (const v8h*)(Kc + ((size_t)t * 32 + b) * 512 + h * 64);
        const v8h* vp8 = (const v8h*)(Vc + ((size_t)t * 32 + b) * 512 + h * 64);
        __builtin_prefetch((const _Float16*)kp8 + 32 * 512, 0, 1);
        __builtin_prefetch((const _Float16*)vp8 + 32 * 512, 0, 1);
        float s = 0.f;
#pragma unroll
        for (int c = 0; c < 8; ++c) {
          v8h kk = kp8[c];
#pragma unroll
          for (int j = 0; j < 8; ++j) s += q[c * 8 + j] * (float)kk[j];
        }
        s *= QK_SCALE;
        float mn   = fmaxf(m, s);
        float corr = __expf(m - mn);
        float w    = __expf(s - mn);
        l = l * corr + w;
#pragma unroll
        for (int c = 0; c < 8; ++c) {
          v8h vv = vp8[c];
#pragma unroll
          for (int j = 0; j < 8; ++j)
            acc[c * 8 + j] = acc[c * 8 + j] * corr + w * (float)vv[j];
        }
        m = mn;
      }
      float rl = 1.f / l;
      v8h* o8 = (v8h*)(qhbuf + b * 512 + h * 64);      // head_out overwrites qh
#pragma unroll
      for (int c = 0; c < 8; ++c) {
        v8h o;
#pragma unroll
        for (int j = 0; j < 8; ++j) o[j] = (_Float16)(acc[c * 8 + j] * rl);
        o8[c] = o;
      }
    }
    __syncthreads();

    // GEMM3: attended = head_out @ Wo.T -> actbuf[:, 1024:1536]
    wave_gemm_f16(qhbuf, 512, BIGSPLIT, 0, woP, 32, 16, nullptr,
                  actbuf + 1024, 2048, wave, NW, lane);
    __syncthreads();

    // GEMM4: inter_in[32x2048] @ int_W.T -> raw interbuf, then LN+tanh in place
    wave_gemm_f16(actbuf, 2048, BIGSPLIT, 0, intWp, 128, 64, int_b,
                  interbuf, 2048, wave, NW, lane);
    __syncthreads();
    ln_tanh(interbuf, 2048, 2048, in_g, in_b, interbuf, 2048, pred, stats, tid);

    // GEMM5: inter @ fin_W.T -> raw actbuf[:, 0:1536], then LN+tanh in place
    wave_gemm_f16(interbuf, 2048, BIGSPLIT, 0, finWp, 96, 64, fin_b,
                  actbuf, 2048, wave, NW, lane);
    __syncthreads();
    ln_tanh(actbuf, 2048, 1536, fn_g, fn_b, actbuf, 2048, pred, stats, tid);

    // h_f = fin[:, 1024:1536] -> prev_h slot + global hs (128-bit moves)
    for (int idx = tid; idx < 32 * 64; idx += 512) {
      int b = idx >> 6, c8 = idx & 63;
      v8h v = *(const v8h*)(actbuf + b * 2048 + 1024 + c8 * 8);
      *(v8h*)(actbuf + b * 2048 + 1536 + c8 * 8) = v;
      *(v8h*)(hs + ((size_t)i * 32 + b) * 512 + c8 * 8) = v;
    }
    __syncthreads();
  }
}

// ============================================================================
// Decode: out[8192 x 10001] = hs[8192 x 512](f16) @ dec_W.T + dec_b.
// Block = 128 threads (4 waves) -> one 16-row M-tile x 16 N-tiles.
// A row-block (16x512 f16 = 16KB) staged to LDS via async global->LDS copies;
// each wave keeps 4 N-tile accumulators so A-frags are reused 4x from LDS.
// ============================================================================
__global__ void __launch_bounds__(128) decode_kernel(
    const _Float16* __restrict__ hs,
    const _Float16* __restrict__ decWp,
    const float* __restrict__ dec_b,
    float* __restrict__ out) {
  __shared__ _Float16 atile[16 * 512];         // only LDS object -> offset 0
  const int Mtiles = (T_STEPS * BATCH) / 16;   // 512
  const int Ntiles = NPAD / 16;                // 626
  int tid  = threadIdx.x;
  int wave = tid >> 5;
  int lane = tid & 31;
  int mt   = blockIdx.x % Mtiles;
  int ng   = blockIdx.x / Mtiles;              // 0..39

  // async-stage A block: 16 rows x 512 halves = 1024 x 16B transfers
  const _Float16* abase = hs + (size_t)mt * 16 * 512;
#pragma unroll 2
  for (int c = tid; c < 1024; c += 128)
    async_g2lds_b128((unsigned)(c * 16), abase + (size_t)c * 8);
  wait_async0();
  __syncthreads();

  int ntBase = ng * 16 + wave * 4;
  v8f acc0 = {}, acc1 = {}, acc2 = {}, acc3 = {};
  int rlo = lane & 15;
  int ks  = (lane >> 4) * 8;
  const _Float16* bp = decWp + (size_t)ntBase * 512 + lane * 16;
#pragma unroll 2
  for (int kt = 0; kt < 16; ++kt) {
    v16h a = load_a_frag(atile + rlo * 512 + kt * 32 + ks);
    size_t boff = (size_t)kt * Ntiles * 512;
    if (ntBase + 0 < Ntiles) {
      v16h b = *(const v16h*)(bp + boff);
      acc0 = __builtin_amdgcn_wmma_f32_16x16x32_f16(false, a, false, b, (short)0, acc0, false, false);
    }
    if (ntBase + 1 < Ntiles) {
      v16h b = *(const v16h*)(bp + boff + 512);
      acc1 = __builtin_amdgcn_wmma_f32_16x16x32_f16(false, a, false, b, (short)0, acc1, false, false);
    }
    if (ntBase + 2 < Ntiles) {
      v16h b = *(const v16h*)(bp + boff + 1024);
      acc2 = __builtin_amdgcn_wmma_f32_16x16x32_f16(false, a, false, b, (short)0, acc2, false, false);
    }
    if (ntBase + 3 < Ntiles) {
      v16h b = *(const v16h*)(bp + boff + 1536);
      acc3 = __builtin_amdgcn_wmma_f32_16x16x32_f16(false, a, false, b, (short)0, acc3, false, false);
    }
  }

  int subrow = (lane >= 16) ? 8 : 0;
  v8f accs[4] = {acc0, acc1, acc2, acc3};
#pragma unroll
  for (int q = 0; q < 4; ++q) {
    int nt = ntBase + q;
    if (nt >= Ntiles) break;
    int n = nt * 16 + (lane & 15);
    if (n < VOCAB) {
      float bv = dec_b[n];
#pragma unroll
      for (int v = 0; v < 8; ++v)
        out[(size_t)(mt * 16 + subrow + v) * VOCAB + n] = accs[q][v] + bv;
    }
  }
}

// ============================================================================
extern "C" void kernel_launch(void* const* d_in, const int* in_sizes, int n_in,
                              void* d_out, int out_size, void* d_ws, size_t ws_size,
                              hipStream_t stream) {
  const int*   obs         = (const int*)  d_in[0];
  const float* hidden_init = (const float*)d_in[1];
  const float* emb         = (const float*)d_in[4];
  const float* qW          = (const float*)d_in[5];
  const float* q_b         = (const float*)d_in[6];
  const float* qn_g        = (const float*)d_in[7];
  const float* qn_b        = (const float*)d_in[8];
  const float* Wq          = (const float*)d_in[9];
  const float* Wk          = (const float*)d_in[10];
  const float* Wv          = (const float*)d_in[11];
  const float* Wo          = (const float*)d_in[12];
  const float* intW        = (const float*)d_in[13];
  const float* int_b       = (const float*)d_in[14];
  const float* in_g        = (const float*)d_in[15];
  const float* in_b        = (const float*)d_in[16];
  const float* finW        = (const float*)d_in[17];
  const float* fin_b       = (const float*)d_in[18];
  const float* fn_g        = (const float*)d_in[19];
  const float* fn_b        = (const float*)d_in[20];
  const float* decW        = (const float*)d_in[21];
  const float* dec_b       = (const float*)d_in[22];
  float* out = (float*)d_out;

  // workspace carve-up (halves); all offsets multiples of 512 halves (1KB)
  _Float16* ws = (_Float16*)d_ws;
  size_t off = 0;
  auto carve = [&](size_t halves) { _Float16* p = ws + off; off += halves; return p; };
  _Float16* qWp   = carve((size_t)32 * 32  * 512);   // K=1024, N=512
  _Float16* wqP   = carve((size_t)16 * 32  * 512);   // K=512,  N=512
  _Float16* wkP   = carve((size_t)16 * 32  * 512);
  _Float16* wvP   = carve((size_t)16 * 32  * 512);
  _Float16* woP   = carve((size_t)16 * 32  * 512);
  _Float16* intWp = carve((size_t)64 * 128 * 512);   // K=2048, N=2048
  _Float16* finWp = carve((size_t)64 * 96  * 512);   // K=2048, N=1536
  _Float16* decWp = carve((size_t)16 * 626 * 512);   // K=512,  N=10016
  _Float16* Kc    = carve((size_t)T_STEPS * BATCH * HID);
  _Float16* Vc    = carve((size_t)T_STEPS * BATCH * HID);
  _Float16* hs    = carve((size_t)T_STEPS * BATCH * HID);

  auto packBlocks = [](int Ntiles, int Ktiles) {
    return (Ntiles * Ktiles * 32 + 255) / 256;
  };
  pack_b_frag<<<packBlocks(32, 32),  256, 0, stream>>>(qW,   qWp,   32,  32, 1024, 512);
  pack_b_frag<<<packBlocks(32, 16),  256, 0, stream>>>(Wq,   wqP,   32,  16, 512,  512);
  pack_b_frag<<<packBlocks(32, 16),  256, 0, stream>>>(Wk,   wkP,   32,  16, 512,  512);
  pack_b_frag<<<packBlocks(32, 16),  256, 0, stream>>>(Wv,   wvP,   32,  16, 512,  512);
  pack_b_frag<<<packBlocks(32, 16),  256, 0, stream>>>(Wo,   woP,   32,  16, 512,  512);
  pack_b_frag<<<packBlocks(128, 64), 256, 0, stream>>>(intW, intWp, 128, 64, 2048, 2048);
  pack_b_frag<<<packBlocks(96, 64),  256, 0, stream>>>(finW, finWp, 96,  64, 2048, 1536);
  pack_b_frag<<<packBlocks(626, 16), 256, 0, stream>>>(decW, decWp, 626, 16, 512,  VOCAB);

  // LDS: act(128K) + inter(128K) + qh(32K) + reduce scratch (~4.3K) ~= 292.5 KB
  size_t smemBytes = (size_t)(32 * 2048 + 32 * 2048 + 32 * 512) * sizeof(_Float16)
                   + (size_t)(32 * 16 * 2 + 32 * 2) * sizeof(float);
  rnn_scan_kernel<<<1, 512, smemBytes, stream>>>(
      obs, hidden_init, emb, qWp, q_b, qn_g, qn_b,
      wqP, wkP, wvP, woP, intWp, int_b, in_g, in_b,
      finWp, fin_b, fn_g, fn_b, Kc, Vc, hs);

  const int Mtiles = (T_STEPS * BATCH) / 16;   // 512
  const int NtGrp  = (NPAD / 16 + 15) / 16;    // 40 groups of 16 N-tiles
  decode_kernel<<<Mtiles * NtGrp, 128, 0, stream>>>(hs, decWp, dec_b, out);
}
